// WeightlessGQA_13967233647269
// MI455X (gfx1250) — compile-verified, matrix operand order checked by reading, not run on hardware
//
#include <hip/hip_runtime.h>
#include <stdint.h>

// ---------------------------------------------------------------------------
// WeightlessGQA for MI455X (gfx1250, wave32, WMMA 16x16x32 bf16)
// bf16x3 split-precision WMMA (~2.7x f32-WMMA rate, ~2^-15 rel error),
// flash attention, all bf16 splitting done ONCE per element via v_perm_b32.
// ---------------------------------------------------------------------------

#define DEVINL __device__ __forceinline__

typedef __bf16 v16bf __attribute__((ext_vector_type(16)));
typedef float  v8f   __attribute__((ext_vector_type(8)));

struct alignas(16) Frag { uint32_t x[8]; };   // 16x32 bf16 tile share (8 VGPRs)

// Truncation split of two consecutive f32 into packed bf16 hi + residual lo.
// hi = [top16(b) : top16(a)] in ONE v_perm_b32; residual captured the same way.
DEVINL void split2(float a, float b, uint32_t& hi, uint32_t& lo) {
  uint32_t ua = __float_as_uint(a), ub = __float_as_uint(b);
  hi = __builtin_amdgcn_perm(ub, ua, 0x07060302u);
  float ra = a - __uint_as_float(ua & 0xffff0000u);
  float rb = b - __uint_as_float(ub & 0xffff0000u);
  lo = __builtin_amdgcn_perm(__float_as_uint(rb), __float_as_uint(ra), 0x07060302u);
}

// D = (Ah+Al)(Bh+Bl) + C ~= Ah*Bh + Ah*Bl + Al*Bh + C  (3x bf16 WMMA)
DEVINL v8f wmma3(const Frag& ah, const Frag& al, const Frag& bh, const Frag& bl, v8f c) {
  v16bf AH = __builtin_bit_cast(v16bf, ah);
  v16bf AL = __builtin_bit_cast(v16bf, al);
  v16bf BH = __builtin_bit_cast(v16bf, bh);
  v16bf BL = __builtin_bit_cast(v16bf, bl);
  c = __builtin_amdgcn_wmma_f32_16x16x32_bf16(false, AH, false, BH, (short)0, c, false, false);
  c = __builtin_amdgcn_wmma_f32_16x16x32_bf16(false, AH, false, BL, (short)0, c, false, false);
  c = __builtin_amdgcn_wmma_f32_16x16x32_bf16(false, AL, false, BH, (short)0, c, false, false);
  return c;
}

// load packed-pair fragment (A-layout): 2x b128 per plane
DEVINL void load_frag_pk(const uint32_t* __restrict__ base, int hh, Frag& f) {
  *reinterpret_cast<uint4*>(&f.x[0]) = *reinterpret_cast<const uint4*>(base + hh * 4);
  *reinterpret_cast<uint4*>(&f.x[4]) = *reinterpret_cast<const uint4*>(base + 8 + hh * 4);
}

// ---------------------------------------------------------------------------
// GEMM: C[M,N] = A[M,K] @ W[N,K]^T, f32 in, bf16x3 WMMA.
// MODE 0: f32 C stored [M,N].  MODE 1: pack V^T hi/lo planes [b,kv,hd,s/2].
// Block 256 thr (8 waves), tile 128x64, wave tile 32x32, K-step 32.
// ---------------------------------------------------------------------------
template<int MODE>
__global__ __launch_bounds__(256) void gemm_k(const float* __restrict__ A,
                                              const float* __restrict__ W,
                                              float* __restrict__ C,
                                              uint32_t* __restrict__ Phi,
                                              uint32_t* __restrict__ Plo,
                                              int M, int N, int K) {
  constexpr int BM = 128, BN = 64, BK = 32;
  __shared__ uint32_t sAh[BM][20], sAl[BM][20], sBh[BN][20], sBl[BN][20];
  const int tid = threadIdx.x;
  const int m0 = blockIdx.y * BM, n0 = blockIdx.x * BN;
  const int wave = tid >> 5, lane = tid & 31;
  const int wm = wave >> 1, wn = wave & 1;
  const int r = lane & 15, hh = lane >> 4;
  v8f acc[2][2] = {};

  for (int k0 = 0; k0 < K; k0 += BK) {
    #pragma unroll
    for (int i = 0; i < 4; ++i) {               // A tile 128x32
      int idx = tid + 256 * i;
      int row = idx >> 3, f4 = idx & 7;
      const float4 v = *reinterpret_cast<const float4*>(&A[(size_t)(m0 + row) * K + k0 + 4 * f4]);
      uint32_t h0, l0, h1, l1;
      split2(v.x, v.y, h0, l0);
      split2(v.z, v.w, h1, l1);
      sAh[row][2 * f4] = h0; sAh[row][2 * f4 + 1] = h1;
      sAl[row][2 * f4] = l0; sAl[row][2 * f4 + 1] = l1;
    }
    #pragma unroll
    for (int i = 0; i < 2; ++i) {               // B tile 64x32
      int idx = tid + 256 * i;
      int row = idx >> 3, f4 = idx & 7;
      const float4 v = *reinterpret_cast<const float4*>(&W[(size_t)(n0 + row) * K + k0 + 4 * f4]);
      uint32_t h0, l0, h1, l1;
      split2(v.x, v.y, h0, l0);
      split2(v.z, v.w, h1, l1);
      sBh[row][2 * f4] = h0; sBh[row][2 * f4 + 1] = h1;
      sBl[row][2 * f4] = l0; sBl[row][2 * f4 + 1] = l1;
    }
    if (k0 + BK < K) {
      int row = tid >> 3, f4 = tid & 7;
      __builtin_prefetch(&A[(size_t)(m0 + row) * K + k0 + BK + 4 * f4], 0, 3);
      __builtin_prefetch(&W[(size_t)(n0 + row) * K + k0 + BK + 4 * f4], 0, 3);
    }
    __syncthreads();

    Frag ah[2], al[2], bh[2], bl[2];            // 2x ds_load_b128 per plane
    #pragma unroll
    for (int i = 0; i < 2; ++i) {
      int ra = wm * 32 + i * 16 + r;
      int rb = wn * 32 + i * 16 + r;
      load_frag_pk(&sAh[ra][0], hh, ah[i]);
      load_frag_pk(&sAl[ra][0], hh, al[i]);
      load_frag_pk(&sBh[rb][0], hh, bh[i]);
      load_frag_pk(&sBl[rb][0], hh, bl[i]);
    }
    #pragma unroll
    for (int i = 0; i < 2; ++i)
      #pragma unroll
      for (int j = 0; j < 2; ++j)
        acc[i][j] = wmma3(ah[i], al[i], bh[j], bl[j], acc[i][j]);
    __syncthreads();
  }

  // epilogue: C/D layout = lane col (lane&15), rows 8*(lane>>4)+v
  #pragma unroll
  for (int i = 0; i < 2; ++i)
    #pragma unroll
    for (int j = 0; j < 2; ++j) {
      int mg0 = m0 + wm * 32 + i * 16 + 8 * hh;
      int ng  = n0 + wn * 32 + j * 16 + r;
      if (MODE == 0) {
        #pragma unroll
        for (int v = 0; v < 8; ++v)
          C[(size_t)(mg0 + v) * N + ng] = acc[i][j][v];
      } else {
        // pack pairs along m (= sequence t) -> V^T planes [b, kv, hd, s/2]
        uint32_t h[4], l[4];
        #pragma unroll
        for (int vp = 0; vp < 4; ++vp)
          split2(acc[i][j][2 * vp], acc[i][j][2 * vp + 1], h[vp], l[vp]);
        int kvh = ng >> 7, d = ng & 127;
        int bi = mg0 >> 11, t = mg0 & 2047;     // S = 2048
        size_t base = ((size_t)((bi * 8 + kvh) * 128 + d)) * 1024 + (t >> 1);
        *reinterpret_cast<uint4*>(&Phi[base]) = make_uint4(h[0], h[1], h[2], h[3]);
        *reinterpret_cast<uint4*>(&Plo[base]) = make_uint4(l[0], l[1], l[2], l[3]);
      }
    }
}

// ---------------------------------------------------------------------------
// RoPE (Gemma2) + pack: reads f32 Q/K, writes packed bf16 hi/lo pair planes.
// Q: [tok, 16, 64] pairs (1/sqrt(128) folded in).  K: [tok, 8, 64] pairs.
// One thread = one (token, head, pair p): dims {2p,2p+1} and {2p+64,2p+65}.
// ---------------------------------------------------------------------------
__global__ __launch_bounds__(256) void rope_pack_k(const float* __restrict__ Qf,
                                                   const float* __restrict__ Kf,
                                                   uint32_t* __restrict__ Qhi,
                                                   uint32_t* __restrict__ Qlo,
                                                   uint32_t* __restrict__ Khi,
                                                   uint32_t* __restrict__ Klo) {
  const int idx = blockIdx.x * 256 + threadIdx.x;
  const int p = idx & 31;
  int rest = idx >> 5;
  const int head = rest % 24;
  const int tok = rest / 24;                  // bi*2048 + pos
  const int pos = tok & 2047;
  const float inv0 = __powf(10000.0f, -(float)(2 * p)     * (1.0f / 64.0f));
  const float inv1 = __powf(10000.0f, -(float)(2 * p + 1) * (1.0f / 64.0f));
  float s0, c0, s1, c1;
  __sincosf((float)pos * inv0, &s0, &c0);
  __sincosf((float)pos * inv1, &s1, &c1);
  if (head < 16) {
    const float scale = 0.08838834764831845f;  // 128^-0.5
    const float* src = Qf + (size_t)tok * 2048 + head * 128;
    float xa0 = src[2 * p], xa1 = src[2 * p + 1];
    float xb0 = src[2 * p + 64], xb1 = src[2 * p + 65];
    float oa0 = (xa0 * c0 - xb0 * s0) * scale;
    float oa1 = (xa1 * c1 - xb1 * s1) * scale;
    float ob0 = (xb0 * c0 + xa0 * s0) * scale;
    float ob1 = (xb1 * c1 + xa1 * s1) * scale;
    uint32_t h0, l0, h1, l1;
    split2(oa0, oa1, h0, l0);
    split2(ob0, ob1, h1, l1);
    size_t dst = (size_t)tok * 1024 + head * 64;
    Qhi[dst + p] = h0; Qhi[dst + 32 + p] = h1;
    Qlo[dst + p] = l0; Qlo[dst + 32 + p] = l1;
  } else {
    int hk = head - 16;
    const float* src = Kf + (size_t)tok * 1024 + hk * 128;
    float xa0 = src[2 * p], xa1 = src[2 * p + 1];
    float xb0 = src[2 * p + 64], xb1 = src[2 * p + 65];
    float oa0 = xa0 * c0 - xb0 * s0;
    float oa1 = xa1 * c1 - xb1 * s1;
    float ob0 = xb0 * c0 + xa0 * s0;
    float ob1 = xb1 * c1 + xa1 * s1;
    uint32_t h0, l0, h1, l1;
    split2(oa0, oa1, h0, l0);
    split2(ob0, ob1, h1, l1);
    size_t dst = (size_t)tok * 512 + hk * 64;
    Khi[dst + p] = h0; Khi[dst + 32 + p] = h1;
    Klo[dst + p] = l0; Klo[dst + 32 + p] = l1;
  }
}

// ---------------------------------------------------------------------------
// Flash attention: 4 waves/block; each wave owns 16 q-rows of one (b,h),
// streams 32 keys/iter. Q/K/V fragments are pure b128 loads of packed planes.
// ---------------------------------------------------------------------------
__global__ __launch_bounds__(128) void attn_k(const uint32_t* __restrict__ Qhi,
                                              const uint32_t* __restrict__ Qlo,
                                              const uint32_t* __restrict__ Khi,
                                              const uint32_t* __restrict__ Klo,
                                              const uint32_t* __restrict__ Vhi,
                                              const uint32_t* __restrict__ Vlo,
                                              float* __restrict__ O) {
  __shared__ float sP[4][16][34];              // per-wave P transpose buffer
  const int mblk = blockIdx.x;
  const int bi = blockIdx.y >> 4, h = blockIdx.y & 15, kvh = h >> 1;
  const int wv = threadIdx.x >> 5, lane = threadIdx.x & 31;
  const int r = lane & 15, hh = lane >> 4;
  const int m0 = mblk * 64 + wv * 16;

  Frag qh[4], ql[4];
  {
    const size_t qrow = (size_t)(bi * 2048 + m0 + r) * 1024 + h * 64;
    #pragma unroll
    for (int ck = 0; ck < 4; ++ck) {
      load_frag_pk(Qhi + qrow + ck * 16, hh, qh[ck]);
      load_frag_pk(Qlo + qrow + ck * 16, hh, ql[ck]);
    }
  }

  float Mst[8], Lst[8];
  #pragma unroll
  for (int v = 0; v < 8; ++v) { Mst[v] = -1e30f; Lst[v] = 0.0f; }
  v8f Ot[8] = {};

  const int tEnd = mblk * 64 + 64;             // uniform per block
  for (int t0 = 0; t0 < tEnd; t0 += 32) {
    v8f s0 = {}, s1 = {};
    #pragma unroll
    for (int ck = 0; ck < 4; ++ck) {
      Frag kh, kl;
      const size_t krow = (size_t)(bi * 2048 + t0 + r) * 512 + kvh * 64 + ck * 16;
      load_frag_pk(Khi + krow, hh, kh);
      load_frag_pk(Klo + krow, hh, kl);
      s0 = wmma3(qh[ck], ql[ck], kh, kl, s0);
      load_frag_pk(Khi + krow + (size_t)16 * 512, hh, kh);
      load_frag_pk(Klo + krow + (size_t)16 * 512, hh, kl);
      s1 = wmma3(qh[ck], ql[ck], kh, kl, s1);
    }
    // softcap + causal + online softmax (row m = 8*hh+v across 16 lanes)
    #pragma unroll
    for (int v = 0; v < 8; ++v) {
      int qg = m0 + 8 * hh + v;
      float a = 50.0f * tanhf(s0[v] * 0.02f);
      if (t0 + r > qg)      a = -2.3819763e38f;
      float b = 50.0f * tanhf(s1[v] * 0.02f);
      if (t0 + 16 + r > qg) b = -2.3819763e38f;
      float mx = fmaxf(a, b);
      #pragma unroll
      for (int msk = 1; msk <= 8; msk <<= 1) mx = fmaxf(mx, __shfl_xor(mx, msk, 32));
      float Mn = fmaxf(Mst[v], mx);
      float f  = __expf(Mst[v] - Mn);
      a = __expf(a - Mn); b = __expf(b - Mn);
      float rs = a + b;
      #pragma unroll
      for (int msk = 1; msk <= 8; msk <<= 1) rs += __shfl_xor(rs, msk, 32);
      Lst[v] = Lst[v] * f + rs;
      Mst[v] = Mn;
      #pragma unroll
      for (int dt = 0; dt < 8; ++dt) Ot[dt][v] *= f;
      sP[wv][8 * hh + v][r]      = a;
      sP[wv][8 * hh + v][16 + r] = b;
    }
    __syncthreads();
    Frag ph, pl;
    #pragma unroll
    for (int j = 0; j < 8; ++j) {
      int cc = ((j >> 2) << 4) + (hh << 3) + ((j & 3) << 1);
      split2(sP[wv][r][cc], sP[wv][r][cc + 1], ph.x[j], pl.x[j]);
    }
    #pragma unroll
    for (int dt = 0; dt < 8; ++dt) {
      Frag vh, vl;
      const size_t vrow = ((size_t)((bi * 8 + kvh) * 128 + dt * 16 + r)) * 1024 + (t0 >> 1);
      load_frag_pk(Vhi + vrow, hh, vh);
      load_frag_pk(Vlo + vrow, hh, vl);
      Ot[dt] = wmma3(ph, pl, vh, vl, Ot[dt]);
    }
    __syncthreads();
  }

  #pragma unroll
  for (int v = 0; v < 8; ++v) {
    int qg = m0 + 8 * hh + v;
    float inv = 1.0f / Lst[v];
    #pragma unroll
    for (int dt = 0; dt < 8; ++dt)
      O[((size_t)(bi * 2048 + qg)) * 2048 + h * 128 + dt * 16 + r] = Ot[dt][v] * inv;
  }
}

// ---------------------------------------------------------------------------
extern "C" void kernel_launch(void* const* d_in, const int* in_sizes, int n_in,
                              void* d_out, int out_size, void* d_ws, size_t ws_size,
                              hipStream_t stream) {
  (void)in_sizes; (void)n_in; (void)out_size; (void)ws_size;
  const float* x  = (const float*)d_in[0];
  const float* wq = (const float*)d_in[1];
  const float* wk = (const float*)d_in[2];
  const float* wv = (const float*)d_in[3];
  const float* wo = (const float*)d_in[4];
  float* out = (float*)d_out;
  float* wsf = (float*)d_ws;
  uint32_t* wsu = (uint32_t*)d_ws;

  const int M = 4096;                                   // b*s
  // dword-offset workspace map (~117 MB total)
  float*    Qf  = wsf;                                  // [4096,2048] f32; reused as AO
  float*    Kf  = wsf + (size_t)8388608;                // [4096,1024] f32
  uint32_t* Qhi = wsu + (size_t)12582912;               // [4096,16,64] pairs
  uint32_t* Qlo = wsu + (size_t)16777216;
  uint32_t* Khi = wsu + (size_t)20971520;               // [4096,8,64]
  uint32_t* Klo = wsu + (size_t)23068672;
  uint32_t* Vhi = wsu + (size_t)25165824;               // [2,8,128,1024]
  uint32_t* Vlo = wsu + (size_t)27262976;
  float*    AO  = Qf;                                   // alias (Qf dead after rope)

  dim3 blk(256);
  gemm_k<0><<<dim3(2048 / 64, 4096 / 128), blk, 0, stream>>>(x, wq, Qf, nullptr, nullptr, M, 2048, 2048);
  gemm_k<0><<<dim3(1024 / 64, 4096 / 128), blk, 0, stream>>>(x, wk, Kf, nullptr, nullptr, M, 1024, 2048);
  gemm_k<1><<<dim3(1024 / 64, 4096 / 128), blk, 0, stream>>>(x, wv, nullptr, Vhi, Vlo, M, 1024, 2048);
  rope_pack_k<<<dim3((4096 * 24 * 32) / 256), blk, 0, stream>>>(Qf, Kf, Qhi, Qlo, Khi, Klo);
  attn_k<<<dim3(32, 32), dim3(128), 0, stream>>>(Qhi, Qlo, Khi, Klo, Vhi, Vlo, AO);
  gemm_k<0><<<dim3(2048 / 64, 4096 / 128), blk, 0, stream>>>(AO, wo, out, nullptr, nullptr, M, 2048, 2048);
}
